// TransBlock_39135742001460
// MI455X (gfx1250) — matrix operand order repaired and checked by reading, not compiled, not run on hardware
//
#include <hip/hip_runtime.h>
#include <hip/hip_bf16.h>
#include <math.h>
#include <cstdint>

typedef _Float16 h16;
typedef __attribute__((ext_vector_type(16))) _Float16 v16h;
typedef __attribute__((ext_vector_type(8)))  _Float16 v8h;
typedef __attribute__((ext_vector_type(8)))  float    v8f;

#define DI static __device__ __forceinline__

namespace {
constexpr int BATCH = 2;
constexpr int SEQ   = 4096;
constexpr int DH    = 64;    // head dim == model dim
constexpr int NH    = 8;     // heads
constexpr int HD    = 512;   // NH * DH
constexpr int DFF   = 256;
constexpr int QKVC  = 1536;  // 3 * HD
constexpr int ROWS  = BATCH * SEQ; // 8192

// workspace byte offsets (all 256-aligned)
constexpr size_t OFF_QH   = 0;                               // [16][4096][64] f16, pre-scaled
constexpr size_t OFF_KH   = OFF_QH  + (size_t)16*SEQ*DH*2;   // 8 MiB
constexpr size_t OFF_VT   = OFF_KH  + (size_t)16*SEQ*DH*2;   // V transposed [16][64][4096] f16
constexpr size_t OFF_AV   = OFF_VT  + (size_t)16*DH*SEQ*2;   // [8192][512] f16
constexpr size_t OFF_RES  = OFF_AV  + (size_t)ROWS*HD*2;     // [8192][64] f32
constexpr size_t OFF_WQKT = OFF_RES + (size_t)ROWS*DH*4;     // [1536][64] f16
constexpr size_t OFF_WOT  = OFF_WQKT + (size_t)QKVC*DH*2;    // [64][512] f16
constexpr size_t OFF_W1T  = OFF_WOT + (size_t)DH*HD*2;       // [256][64] f16
constexpr size_t OFF_W2T  = OFF_W1T + (size_t)DFF*DH*2;      // [64][256] f16
}

// ---- gfx1250 async global->LDS path via inline asm (VGLOBAL GV mode) ----
// VDST VGPR = LDS byte offset (generic-pointer low 32 bits == wave LDS offset
// per the flat-aperture rule), VADDR pair = 64-bit global address.
DI void stage_b128(const h16* g, h16* l) {
  uint32_t lds_off = (uint32_t)(uintptr_t)l;
  asm volatile("global_load_async_to_lds_b128 %0, %1, off"
               :
               : "v"(lds_off), "v"(g)
               : "memory");
}

// s_wait_asynccnt N (SOPP); async ops complete in order per wave.
#define WAIT_ASYNC(n) asm volatile("s_wait_asynccnt " #n ::: "memory")

DI v8f wmma_f16(v16h a, v16h b, v8f c) {
  return __builtin_amdgcn_wmma_f32_16x16x32_f16(false, a, false, b, (short)0, c, false, false);
}

// A fragment (16x32, f16) from row-major [rows][ld] at tile (m0, k0).
// Lane half h=(l>>4), row m=l&15; regs 0..3 hold K=h*8+0..7, regs 4..7 hold K=16+h*8+0..7.
DI v16h load_a_frag(const h16* base, int ld, int m0, int k0, int lane) {
  int m = lane & 15, half = lane >> 4;
  const h16* p = base + (size_t)(m0 + m) * ld + k0 + half * 8;
  v8h lo = *(const v8h*)p;
  v8h hi = *(const v8h*)(p + 16);
  v16h a;
#pragma unroll
  for (int i = 0; i < 8; ++i) { a[i] = lo[i]; a[8 + i] = hi[i]; }
  return a;
}

// B fragment (32x16, f16) from B^T stored row-major [n][ld] (rows are N, contiguous in K).
// Lane half h=(l>>4) holds K = h*8 + 0..7 and 16 + h*8 + 0..7, col n = l&15. Two 16B loads (16B-aligned strides).
DI v16h load_b_fragT(const h16* bt, int ld, int n0, int k0, int lane) {
  int n = lane & 15, half = lane >> 4;
  const h16* p = bt + (size_t)(n0 + n) * ld + k0 + half * 8;
  v8h lo = *(const v8h*)p;
  v8h hi = *(const v8h*)(p + 16);
  v16h b;
#pragma unroll
  for (int i = 0; i < 8; ++i) { b[i] = lo[i]; b[8 + i] = hi[i]; }
  return b;
}

// ---------------- kernel 0: weight convert + transpose to f16 ----------------
__global__ void k_prep_weights(const float* __restrict__ Wqkv, const float* __restrict__ Wo,
                               const float* __restrict__ W1, const float* __restrict__ W2,
                               h16* wqkT, h16* woT, h16* w1T, h16* w2T) {
  int i = blockIdx.x * blockDim.x + threadIdx.x;
  int stride = gridDim.x * blockDim.x;
  for (int t = i; t < QKVC * DH; t += stride) { int n = t / DH, k = t % DH; wqkT[t] = (h16)Wqkv[k * QKVC + n]; }
  for (int t = i; t < DH * HD;   t += stride) { int n = t / HD, k = t % HD; woT[t]  = (h16)Wo[k * DH + n]; }
  for (int t = i; t < DFF * DH;  t += stride) { int n = t / DH, k = t % DH; w1T[t]  = (h16)W1[k * DFF + n]; }
  for (int t = i; t < DH * DFF;  t += stride) { int n = t / DFF, k = t % DFF; w2T[t] = (h16)W2[k * DH + n]; }
}

// ---------------- kernel 1: LN1 + QKV projection ----------------
__global__ __launch_bounds__(256) void k_ln1_qkv(const float* __restrict__ x,
                                                 const float* __restrict__ lnw,
                                                 const float* __restrict__ lnb,
                                                 const h16* __restrict__ wqkT,
                                                 const float* __restrict__ bqkv,
                                                 h16* __restrict__ qh, h16* __restrict__ kh,
                                                 h16* __restrict__ vt) {
  __shared__ h16 xn[64 * 72];
  int tid = threadIdx.x;
  int row0 = blockIdx.x * 64;
  __builtin_prefetch(wqkT + (size_t)tid * 384, 0, 3);  // warm L2 (global_prefetch_b8)
  if (tid < 64) {
    const float* xr = x + (size_t)(row0 + tid) * DH;
    float mu = 0.f;
#pragma unroll
    for (int k = 0; k < DH; ++k) mu += xr[k];
    mu *= (1.0f / DH);
    float var = 0.f;
#pragma unroll
    for (int k = 0; k < DH; ++k) { float d = xr[k] - mu; var += d * d; }
    var *= (1.0f / DH);
    float rs = rsqrtf(var + 1e-5f);
#pragma unroll
    for (int k = 0; k < DH; ++k)
      xn[tid * 72 + k] = (h16)((xr[k] - mu) * rs * lnw[k] + lnb[k]);
  }
  __syncthreads();
  int wave = tid >> 5, lane = tid & 31;
  int half = lane >> 4, nlane = lane & 15;
  int bb = row0 >> 12;                    // batch (64-row blocks never straddle)
  int tok0 = row0 & (SEQ - 1);
  for (int t = wave; t < 4 * 96; t += 8) {   // M-tiles 4 x N-tiles 96
    int tw = __builtin_amdgcn_readfirstlane(t);  // wave-uniform -> scalar branches
    int mt = tw / 96, nt = tw % 96;
    v16h a0 = load_a_frag(xn, 72, mt * 16, 0, lane);
    v16h a1 = load_a_frag(xn, 72, mt * 16, 32, lane);
    v16h b0 = load_b_fragT(wqkT, DH, nt * 16, 0, lane);
    v16h b1 = load_b_fragT(wqkT, DH, nt * 16, 32, lane);
    v8f c = {};
    c = wmma_f16(a0, b0, c);
    c = wmma_f16(a1, b1, c);
    // 16-wide tile lies inside one section & one head: all uniform
    int sec   = nt >> 5;            // 0=Q 1=K 2=V
    int head  = (nt >> 2) & 7;
    int dcol  = ((nt & 3) << 4) + nlane;
    float bias = bqkv[nt * 16 + nlane];
    size_t bh = (size_t)(bb * NH + head);
    int tokb = tok0 + mt * 16 + 8 * half;
    if (sec == 0) {
      h16* dst = qh + (bh * SEQ + tokb) * DH + dcol;
#pragma unroll
      for (int r = 0; r < 8; ++r) dst[(size_t)r * DH] = (h16)((c[r] + bias) * 0.125f);
    } else if (sec == 1) {
      h16* dst = kh + (bh * SEQ + tokb) * DH + dcol;
#pragma unroll
      for (int r = 0; r < 8; ++r) dst[(size_t)r * DH] = (h16)(c[r] + bias);
    } else {
      h16* dst = vt + (bh * DH + dcol) * SEQ + tokb;
#pragma unroll
      for (int r = 0; r < 8; ++r) dst[r] = (h16)(c[r] + bias);
    }
  }
}

// ---------------- kernel 2: flash attention ----------------
// grid: (b*h)*32 blocks; 8 waves; wave owns 16 query rows; Tc=32 keys/step.
// K/V tiles staged once per block into LDS via async global->LDS, double-buffered.
__global__ __launch_bounds__(256) void k_attn(const h16* __restrict__ qh,
                                              const h16* __restrict__ kh,
                                              const h16* __restrict__ vt,
                                              h16* __restrict__ av) {
  __shared__ h16 kbuf[2][32 * 72];   // [token][dh], padded
  __shared__ h16 vbuf[2][64 * 40];   // [dh][token], padded
  __shared__ h16 plds[8][16 * 40];   // per-wave P re-layout
  int tid = threadIdx.x;
  int wave = tid >> 5, lane = tid & 31;
  int half = lane >> 4, nlane = lane & 15;
  int blk = blockIdx.x;
  int qt = blk & 31;          // 32 q-tiles of 128 rows
  int bh = blk >> 5;          // batch*head
  const h16* Q  = qh + (size_t)bh * SEQ * DH;
  const h16* K  = kh + (size_t)bh * SEQ * DH;
  const h16* Vt = vt + (size_t)bh * DH * SEQ;
  int q0 = qt * 128 + wave * 16;

  v16h qa0 = load_a_frag(Q, DH, q0, 0, lane);
  v16h qa1 = load_a_frag(Q, DH, q0, 32, lane);

  // cooperative staging map: 256 threads x 16B
  int kr = tid >> 3, kc = (tid & 7) * 8;   // K tile: 32 rows x 64
  int vr = tid >> 2, vc = (tid & 3) * 8;   // V tile: 64 rows x 32

  float mrun[8], lrun[8];
  v8f o0 = {}, o1 = {}, o2 = {}, o3 = {};
#pragma unroll
  for (int r = 0; r < 8; ++r) { mrun[r] = -INFINITY; lrun[r] = 0.f; }

  // prologue: stage tile 0 into buffer 0
  stage_b128(K + (size_t)kr * DH + kc,   &kbuf[0][kr * 72 + kc]);
  stage_b128(Vt + (size_t)vr * SEQ + vc, &vbuf[0][vr * 40 + vc]);

  h16* P = &plds[wave][0];
  for (int it = 0; it < SEQ / 32; ++it) {
    int kt = it * 32;
    if (it + 1 < SEQ / 32) {   // stage next tile while computing this one
      int nb = (it + 1) & 1;
      stage_b128(K + (size_t)(kt + 32 + kr) * DH + kc,   &kbuf[nb][kr * 72 + kc]);
      stage_b128(Vt + (size_t)vr * SEQ + (kt + 32) + vc, &vbuf[nb][vr * 40 + vc]);
      WAIT_ASYNC(2);           // retire oldest batch (async ops complete in order)
    } else {
      WAIT_ASYNC(0);
    }
    __syncthreads();           // current buffer fully resident for all waves

    const h16* kb = &kbuf[it & 1][0];
    const h16* vb = &vbuf[it & 1][0];
    v8f s0 = {}, s1 = {};
    s0 = wmma_f16(qa0, load_b_fragT(kb, 72, 0, 0, lane), s0);
    s0 = wmma_f16(qa1, load_b_fragT(kb, 72, 0, 32, lane), s0);
    s1 = wmma_f16(qa0, load_b_fragT(kb, 72, 16, 0, lane), s1);
    s1 = wmma_f16(qa1, load_b_fragT(kb, 72, 16, 32, lane), s1);

#pragma unroll
    for (int r = 0; r < 8; ++r) {
      float rm = fmaxf(s0[r], s1[r]);
      rm = fmaxf(rm, __shfl_xor(rm, 1));
      rm = fmaxf(rm, __shfl_xor(rm, 2));
      rm = fmaxf(rm, __shfl_xor(rm, 4));
      rm = fmaxf(rm, __shfl_xor(rm, 8));     // row max over the 16 n-lanes
      float mnew = fmaxf(mrun[r], rm);
      float alpha = __expf(mrun[r] - mnew);
      mrun[r] = mnew;
      float p0 = __expf(s0[r] - mnew);
      float p1 = __expf(s1[r] - mnew);
      float rs = p0 + p1;
      rs += __shfl_xor(rs, 1);
      rs += __shfl_xor(rs, 2);
      rs += __shfl_xor(rs, 4);
      rs += __shfl_xor(rs, 8);
      lrun[r] = lrun[r] * alpha + rs;
      o0[r] *= alpha; o1[r] *= alpha; o2[r] *= alpha; o3[r] *= alpha;
      int m = r + 8 * half;                  // C-layout row
      P[m * 40 + nlane]      = (h16)p0;
      P[m * 40 + 16 + nlane] = (h16)p1;
    }
    __syncthreads();           // P visible in A-layout order

    v16h pa = load_a_frag(P, 40, 0, 0, lane);
    o0 = wmma_f16(pa, load_b_fragT(vb, 40, 0,  0, lane), o0);
    o1 = wmma_f16(pa, load_b_fragT(vb, 40, 16, 0, lane), o1);
    o2 = wmma_f16(pa, load_b_fragT(vb, 40, 32, 0, lane), o2);
    o3 = wmma_f16(pa, load_b_fragT(vb, 40, 48, 0, lane), o3);
    __syncthreads();           // all reads of this buffer done before re-staging
  }

  int b_ = bh >> 3, hh = bh & 7;
#pragma unroll
  for (int r = 0; r < 8; ++r) {
    float inv = 1.0f / lrun[r];
    int tok = q0 + r + 8 * half;
    h16* dst = av + ((size_t)(b_ * SEQ + tok)) * HD + hh * DH;
    dst[nlane]      = (h16)(o0[r] * inv);
    dst[16 + nlane] = (h16)(o1[r] * inv);
    dst[32 + nlane] = (h16)(o2[r] * inv);
    dst[48 + nlane] = (h16)(o3[r] * inv);
  }
}

// ---------------- kernel 3: Wo projection + residual ----------------
__global__ __launch_bounds__(256) void k_wo_res(const h16* __restrict__ av,
                                                const h16* __restrict__ woT,
                                                const float* __restrict__ bo,
                                                const float* __restrict__ x,
                                                float* __restrict__ res) {
  int wave = threadIdx.x >> 5, lane = threadIdx.x & 31;
  int half = lane >> 4, nlane = lane & 15;
  int row0 = blockIdx.x * 128;     // 64 blocks
#pragma unroll
  for (int i = 0; i < 4; ++i) {    // 32 tiles (M8 x N4), 4 per wave
    int t = wave * 4 + i;
    int mt = t >> 2, nt = t & 3;
    int r0 = row0 + mt * 16;
    v8f c = {};
#pragma unroll
    for (int kc = 0; kc < 16; ++kc) {  // K = 512
      v16h a = load_a_frag(av, HD, r0, kc * 32, lane);
      v16h b = load_b_fragT(woT, HD, nt * 16, kc * 32, lane);
      c = wmma_f16(a, b, c);
    }
    int col = nt * 16 + nlane;
    float bias = bo[col];
#pragma unroll
    for (int r = 0; r < 8; ++r) {
      int gr = r0 + r + 8 * half;
      res[(size_t)gr * DH + col] = c[r] + bias + x[(size_t)gr * DH + col];
    }
  }
}

// ---------------- kernel 4: LN2 + MLP + residual ----------------
__global__ __launch_bounds__(256) void k_mlp(const float* __restrict__ res,
                                             const float* __restrict__ lnw,
                                             const float* __restrict__ lnb,
                                             const h16* __restrict__ w1T,
                                             const float* __restrict__ b1,
                                             const h16* __restrict__ w2T,
                                             const float* __restrict__ b2,
                                             float* __restrict__ out) {
  __shared__ h16 rn[64 * 72];
  __shared__ h16 hact[64 * 264];
  int tid = threadIdx.x;
  int row0 = blockIdx.x * 64;      // 128 blocks
  if (tid < 64) {
    const float* rr = res + (size_t)(row0 + tid) * DH;
    float mu = 0.f;
#pragma unroll
    for (int k = 0; k < DH; ++k) mu += rr[k];
    mu *= (1.0f / DH);
    float var = 0.f;
#pragma unroll
    for (int k = 0; k < DH; ++k) { float d = rr[k] - mu; var += d * d; }
    var *= (1.0f / DH);
    float rs = rsqrtf(var + 1e-5f);
#pragma unroll
    for (int k = 0; k < DH; ++k)
      rn[tid * 72 + k] = (h16)((rr[k] - mu) * rs * lnw[k] + lnb[k]);
  }
  __syncthreads();
  int wave = tid >> 5, lane = tid & 31;
  int half = lane >> 4, nlane = lane & 15;
  // GEMM1: [64x64]@[64x256] -> relu -> LDS
  for (int t = wave; t < 64; t += 8) {     // M4 x N16
    int mt = t >> 4, nt = t & 15;
    v8f c = {};
    c = wmma_f16(load_a_frag(rn, 72, mt * 16, 0, lane),  load_b_fragT(w1T, DH, nt * 16, 0, lane),  c);
    c = wmma_f16(load_a_frag(rn, 72, mt * 16, 32, lane), load_b_fragT(w1T, DH, nt * 16, 32, lane), c);
    int col = nt * 16 + nlane;
    float bias = b1[col];
#pragma unroll
    for (int r = 0; r < 8; ++r) {
      int m = mt * 16 + r + 8 * half;
      hact[m * 264 + col] = (h16)fmaxf(c[r] + bias, 0.0f);
    }
  }
  __syncthreads();
  // GEMM2: [64x256]@[256x64] + b2 + res -> out
  for (int t = wave; t < 16; t += 8) {     // M4 x N4
    int mt = t >> 2, nt = t & 3;
    v8f c = {};
#pragma unroll
    for (int kc = 0; kc < 8; ++kc)
      c = wmma_f16(load_a_frag(hact, 264, mt * 16, kc * 32, lane),
                   load_b_fragT(w2T, DFF, nt * 16, kc * 32, lane), c);
    int col = nt * 16 + nlane;
    float bias = b2[col];
#pragma unroll
    for (int r = 0; r < 8; ++r) {
      int gr = row0 + mt * 16 + r + 8 * half;
      out[(size_t)gr * DH + col] = c[r] + bias + res[(size_t)gr * DH + col];
    }
  }
}

extern "C" void kernel_launch(void* const* d_in, const int* in_sizes, int n_in,
                              void* d_out, int out_size, void* d_ws, size_t ws_size,
                              hipStream_t stream) {
  const float* x     = (const float*)d_in[0];
  const float* ln1_w = (const float*)d_in[1];
  const float* ln1_b = (const float*)d_in[2];
  const float* ln2_w = (const float*)d_in[3];
  const float* ln2_b = (const float*)d_in[4];
  const float* Wqkv  = (const float*)d_in[5];
  const float* bqkv  = (const float*)d_in[6];
  const float* Wo    = (const float*)d_in[7];
  const float* bo    = (const float*)d_in[8];
  const float* W1    = (const float*)d_in[9];
  const float* b1    = (const float*)d_in[10];
  const float* W2    = (const float*)d_in[11];
  const float* b2    = (const float*)d_in[12];
  float* out = (float*)d_out;

  char* ws = (char*)d_ws;
  h16* qh    = (h16*)(ws + OFF_QH);
  h16* kh    = (h16*)(ws + OFF_KH);
  h16* vt    = (h16*)(ws + OFF_VT);
  h16* av    = (h16*)(ws + OFF_AV);
  float* res = (float*)(ws + OFF_RES);
  h16* wqkT  = (h16*)(ws + OFF_WQKT);
  h16* woT   = (h16*)(ws + OFF_WOT);
  h16* w1T   = (h16*)(ws + OFF_W1T);
  h16* w2T   = (h16*)(ws + OFF_W2T);

  k_prep_weights<<<256, 256, 0, stream>>>(Wqkv, Wo, W1, W2, wqkT, woT, w1T, w2T);
  k_ln1_qkv<<<ROWS / 64, 256, 0, stream>>>(x, ln1_w, ln1_b, wqkT, bqkv, qh, kh, vt);
  k_attn<<<16 * 32, 256, 0, stream>>>(qh, kh, vt, av);
  k_wo_res<<<ROWS / 128, 256, 0, stream>>>(av, woT, bo, x, res);
  k_mlp<<<ROWS / 64, 256, 0, stream>>>(res, ln2_w, ln2_b, w1T, b1, w2T, b2, out);
}
